// T1SepClassifier_15693810500346
// MI455X (gfx1250) — matrix operand order repaired and checked by reading, not compiled, not run on hardware
//
#include <hip/hip_runtime.h>
#include <hip/hip_fp16.h>

typedef __attribute__((ext_vector_type(16))) _Float16 v16h;
typedef __attribute__((ext_vector_type(8)))  _Float16 v8h;
typedef __attribute__((ext_vector_type(4)))  _Float16 v4h;
typedef __attribute__((ext_vector_type(8)))  float    v8f;

#define B_TOTAL   262144
#define NWAVES    8
#define TPB       256
#define ROWS_PER_BLOCK (16 * NWAVES)

// ---------------- LDS layout (byte offsets, all 16B aligned) ----------------
// Weight images are stored PRE-SWIZZLED in B-fragment order:
//   half index = ((frag * 32 + lane) * 16 + i), value = W[k = kt*32 + (lane>=16?16:0) + i][n = nt*16 + (lane&15)]
#define OFF_WB1   0         // 3 branches * 6 frags * 512 halves = 18432 B (layer1, Kpad 96, N 32; frag = kt*2+nt)
#define OFF_WB2   18432     // 3 * 5 frags * 512 halves = 15360 B (scores, K 32, Npad 80; frag = nt)
#define OFF_WS1   33792     // 1 frag = 1024 B (sub layer1, Kpad 32, Npad 16)
#define OFF_WS2   34816     // 1 frag = 1024 B (sub scores, Kpad 32, N 16)
#define OFF_WF1   35840     // 16 frags = 16384 B (fc1, Kpad 32, N 256; frag = nt)
#define OFF_BB1   52224     // 3*32 f32
#define OFF_BB2   52608     // 3*80 f32 (pad 0)
#define OFF_BS1   53568     // 16 f32 (pad 0)
#define OFF_BS2   53632     // 16 f32
#define OFF_BF1   53696     // 256 f32
#define OFF_WF2   54720     // 256 f32 (fc2 weights, f32)
#define OFF_BF2   55744     // 1 f32 (+pad)
#define OFF_WAVE  55760
#define PW_XTILE  0         // 16 x 96 f16 = 3072 B (row-major, zero-padded cols)
#define PW_HTILE  3072      // 16 x 32 f16 = 1024 B
#define PW_SCORE  4096      // 16 x 80 f32 = 5120 B
#define PW_FEAT   9216      // 16 x 32 f16 = 1024 B
#define PER_WAVE_BYTES 10240
#define SMEM_TOTAL (OFF_WAVE + NWAVES * PER_WAVE_BYTES)   // 137680 B

#define WMMA_F16(A, Bm, Cm) \
  __builtin_amdgcn_wmma_f32_16x16x32_f16(false, (A), false, (Bm), (short)0, (Cm), false, false)

// A fragment (16x32) from a zero-padded row-major f16 LDS tile.
// 16-bit A layout (contiguous k, matches B images): lanes 0-15 hold K kb+0..15, lanes 16-31 hold kb+16..31.
// Each 8-half run is 16 bytes, 16B-aligned -> 2x ds_load_b128, no guards.
__device__ __forceinline__ v16h a_frag(const _Float16* src, int ld, int kb, int lane) {
  int m = lane & 15;
  int koff = (lane >> 4) * 16;
  const _Float16* p = src + m * ld + kb + koff;
  v8h lo = *(const v8h*)(p);
  v8h hi = *(const v8h*)(p + 8);
  return __builtin_shufflevector(lo, hi, 0, 1, 2, 3, 4, 5, 6, 7, 8, 9, 10, 11, 12, 13, 14, 15);
}

// B fragment: one contiguous 32-byte aligned LDS read from the pre-swizzled image.
__device__ __forceinline__ v16h b_frag(const _Float16* wimg, int frag, int lane) {
  return *(const v16h*)(wimg + ((frag * 32 + lane) << 4));
}

// top-k (descending, lowest index wins ties) + softmax + weighted gather (x from global, cache-hot).
__device__ __forceinline__ void topk_softmax(const float* srow, const float* __restrict__ xrow_g,
                                             int dim, int kk,
                                             int* idx_out, float* w_out,
                                             _Float16* featrow, int fbase) {
  float vals[7];
  int ids[7];
  unsigned long long u0 = 0ull;
  unsigned u1 = 0u;
#pragma unroll
  for (int j = 0; j < kk; ++j) {
    float best = -3.402823e38f;
    int bi = 0;
    for (int c = 0; c < dim; ++c) {
      bool used = (c < 64) ? (((u0 >> c) & 1ull) != 0ull) : (((u1 >> (c - 64)) & 1u) != 0u);
      float s = srow[c];
      if (!used && s > best) { best = s; bi = c; }
    }
    if (bi < 64) u0 |= (1ull << bi); else u1 |= (1u << (bi - 64));
    vals[j] = best;
    ids[j] = bi;
  }
  float mx = vals[0];
  float e[7];
  float sum = 0.0f;
#pragma unroll
  for (int j = 0; j < kk; ++j) { e[j] = __expf(vals[j] - mx); sum += e[j]; }
  float inv = 1.0f / sum;
#pragma unroll
  for (int j = 0; j < kk; ++j) {
    float w = e[j] * inv;
    idx_out[j] = ids[j];
    w_out[j] = w;
    featrow[fbase + j] = (_Float16)(xrow_g[ids[j]] * w);
  }
}

__global__ __launch_bounds__(TPB) void t1sep_fused_kernel(
    const float* __restrict__ thick, const float* __restrict__ area,
    const float* __restrict__ vol, const float* __restrict__ sub_vol,
    const float* __restrict__ tw1, const float* __restrict__ tb1,
    const float* __restrict__ tw2, const float* __restrict__ tb2,
    const float* __restrict__ aw1, const float* __restrict__ ab1,
    const float* __restrict__ aw2, const float* __restrict__ ab2,
    const float* __restrict__ vw1, const float* __restrict__ vb1,
    const float* __restrict__ vw2, const float* __restrict__ vb2,
    const float* __restrict__ sw1, const float* __restrict__ sb1,
    const float* __restrict__ sw2, const float* __restrict__ sb2,
    const float* __restrict__ fc1w, const float* __restrict__ fc1b,
    const float* __restrict__ fc2w, const float* __restrict__ fc2b,
    float* __restrict__ dout) {
  extern __shared__ char smem[];

  _Float16* WB1 = (_Float16*)(smem + OFF_WB1);
  _Float16* WB2 = (_Float16*)(smem + OFF_WB2);
  _Float16* WS1 = (_Float16*)(smem + OFF_WS1);
  _Float16* WS2 = (_Float16*)(smem + OFF_WS2);
  _Float16* WF1 = (_Float16*)(smem + OFF_WF1);
  float* BB1 = (float*)(smem + OFF_BB1);
  float* BB2 = (float*)(smem + OFF_BB2);
  float* BS1 = (float*)(smem + OFF_BS1);
  float* BS2 = (float*)(smem + OFF_BS2);
  float* BF1 = (float*)(smem + OFF_BF1);
  float* WF2 = (float*)(smem + OFF_WF2);
  float* BF2 = (float*)(smem + OFF_BF2);

  const int tid = threadIdx.x;
  const float* W1g[3] = {tw1, aw1, vw1};
  const float* W2g[3] = {tw2, aw2, vw2};
  const float* B1g[3] = {tb1, ab1, vb1};
  const float* B2g[3] = {tb2, ab2, vb2};
  const float* Xg[3] = {thick, area, vol};

  // ---- stage weights into LDS, pre-swizzled into B-fragment order ----
  // WB1: 3 branches x 6 frags (frag = kt*2 + nt), Kpad 96 (68 valid), N 32
  for (int j = tid; j < 3 * 6 * 512; j += TPB) {
    int br = j / 3072, r = j - br * 3072;
    int fr = r >> 9, ln = (r >> 4) & 31, i = r & 15;
    int kt = fr >> 1, nt = fr & 1;
    int k = kt * 32 + ((ln >> 4) << 4) + i;
    int n = nt * 16 + (ln & 15);
    WB1[j] = (k < 68) ? (_Float16)W1g[br][n * 68 + k] : (_Float16)0.0f;
  }
  // WB2: 3 branches x 5 frags (frag = nt), K 32, Npad 80 (68 valid)
  for (int j = tid; j < 3 * 5 * 512; j += TPB) {
    int br = j / 2560, r = j - br * 2560;
    int nt = r >> 9, ln = (r >> 4) & 31, i = r & 15;
    int k = ((ln >> 4) << 4) + i;
    int n = nt * 16 + (ln & 15);
    WB2[j] = (n < 68) ? (_Float16)W2g[br][n * 32 + k] : (_Float16)0.0f;
  }
  // WS1 / WS2: one frag each, Kpad 32, Npad 16
  for (int j = tid; j < 512; j += TPB) {
    int ln = (j >> 4) & 31, i = j & 15;
    int k = ((ln >> 4) << 4) + i;
    int n = ln & 15;
    WS1[j] = (k < 16 && n < 8) ? (_Float16)sw1[n * 16 + k] : (_Float16)0.0f;
    WS2[j] = (k < 8) ? (_Float16)sw2[n * 8 + k] : (_Float16)0.0f;
  }
  // WF1: 16 frags (frag = nt), Kpad 32 (23 valid), N 256
  for (int j = tid; j < 16 * 512; j += TPB) {
    int nt = j >> 9, ln = (j >> 4) & 31, i = j & 15;
    int k = ((ln >> 4) << 4) + i;
    int n = nt * 16 + (ln & 15);
    WF1[j] = (k < 23) ? (_Float16)fc1w[n * 23 + k] : (_Float16)0.0f;
  }
  for (int i = tid; i < 96; i += TPB) BB1[i] = B1g[i >> 5][i & 31];
  for (int i = tid; i < 240; i += TPB) {
    int br = i / 80, n = i - br * 80;
    BB2[i] = (n < 68) ? B2g[br][n] : 0.0f;
  }
  for (int i = tid; i < 16; i += TPB) {
    BS1[i] = (i < 8) ? sb1[i] : 0.0f;
    BS2[i] = sb2[i];
  }
  for (int i = tid; i < 256; i += TPB) {
    BF1[i] = fc1b[i];
    WF2[i] = fc2w[i];
  }
  if (tid == 0) BF2[0] = fc2b[0];
  __syncthreads();

  // ---- output pointers (tuple concatenated flat in return order) ----
  float* out_p = dout;
  float* x1_p = dout + (size_t)B_TOTAL;
  int* tI = (int*)(x1_p + (size_t)B_TOTAL * 256);
  float* tW = (float*)(tI + (size_t)B_TOTAL * 7);
  int* aI = (int*)(tW + (size_t)B_TOTAL * 7);
  float* aW = (float*)(aI + (size_t)B_TOTAL * 7);
  int* vI = (int*)(aW + (size_t)B_TOTAL * 7);
  float* vW = (float*)(vI + (size_t)B_TOTAL * 7);
  int* sI = (int*)(vW + (size_t)B_TOTAL * 7);
  float* sW = (float*)(sI + (size_t)B_TOTAL * 2);
  int* IdxOut[3] = {tI, aI, vI};
  float* WOut[3] = {tW, aW, vW};

  const int lane = tid & 31;
  const int wv = tid >> 5;
  const size_t row0 = (size_t)(blockIdx.x * NWAVES + wv) * 16;

  char* wsc = smem + OFF_WAVE + wv * PER_WAVE_BYTES;
  _Float16* xtile = (_Float16*)(wsc + PW_XTILE);   // 16 x 96 f16
  _Float16* htile = (_Float16*)(wsc + PW_HTILE);   // 16 x 32 f16
  float* scores = (float*)(wsc + PW_SCORE);        // 16 x 80 f32
  _Float16* feat = (_Float16*)(wsc + PW_FEAT);     // 16 x 32 f16

  // zero feature pad + xtile pad columns 68..95 (stay zero across all three branches)
  for (int i = lane; i < 512; i += 32) feat[i] = (_Float16)0.0f;
  for (int i = lane; i < 16 * 14; i += 32) {
    int m = i / 14, c = i - m * 14;
    ((unsigned*)xtile)[m * 48 + 34 + c] = 0u;  // halves [68..95] of row m
  }

  const int n15 = lane & 15;
  const int mo = (lane >> 4) * 8;

  // ================= 68-dim branches =================
  for (int br = 0; br < 3; ++br) {
    const float* xg = Xg[br];
    const _Float16* wb1 = WB1 + br * 3072;
    const _Float16* wb2 = WB2 + br * 2560;
    const float* bb1 = BB1 + br * 32;
    const float* bb2 = BB2 + br * 80;

    // load 16x68 input tile (17 b128 loads per row), convert+pack to f16 rows of 96
    const float4* xg4 = (const float4*)xg;
    for (int i = lane; i < 16 * 17; i += 32) {
      int m = i / 17, c = i - m * 17;
      float4 v = xg4[(row0 + m) * 17 + c];
      v4h h = {(_Float16)v.x, (_Float16)v.y, (_Float16)v.z, (_Float16)v.w};
      *(v4h*)(xtile + m * 96 + c * 4) = h;
    }
    if (br < 2) __builtin_prefetch(&Xg[br + 1][row0 * 68], 0, 1);
    else        __builtin_prefetch(&sub_vol[row0 * 16], 0, 1);

    // layer1: H(16x32) = relu(X(16x96pad) @ W1 + b1); A frags hoisted across n-tiles
    v16h a0 = a_frag(xtile, 96, 0, lane);
    v16h a1 = a_frag(xtile, 96, 32, lane);
    v16h a2 = a_frag(xtile, 96, 64, lane);
#pragma unroll
    for (int nt = 0; nt < 2; ++nt) {
      v8f acc = {};
      acc = WMMA_F16(a0, b_frag(wb1, 0 + nt, lane), acc);
      acc = WMMA_F16(a1, b_frag(wb1, 2 + nt, lane), acc);
      acc = WMMA_F16(a2, b_frag(wb1, 4 + nt, lane), acc);
      int n = nt * 16 + n15;
      float bb = bb1[n];
#pragma unroll
      for (int r = 0; r < 8; ++r)
        htile[(mo + r) * 32 + n] = (_Float16)fmaxf(acc[r] + bb, 0.0f);
    }

    // scores(16x80pad) = H(16x32) @ W2 + b2
    v16h ah = a_frag(htile, 32, 0, lane);
#pragma unroll
    for (int nt = 0; nt < 5; ++nt) {
      v8f c = {};
      c = WMMA_F16(ah, b_frag(wb2, nt, lane), c);
      int n = nt * 16 + n15;
      float bb = bb2[n];
#pragma unroll
      for (int r = 0; r < 8; ++r) scores[(mo + r) * 80 + n] = c[r] + bb;
    }

    // per-sample top-7 + softmax + weighted gather (one sample per lane 0..15)
    if (lane < 16) {
      int m = lane;
      topk_softmax(scores + m * 80, xg + (row0 + m) * 68, 68, 7,
                   IdxOut[br] + (row0 + m) * 7, WOut[br] + (row0 + m) * 7,
                   feat + m * 32, br * 7);
    }
  }

  // ================= sub_vol branch =================
  {
    // zero xtile cols 16..31 (leftover branch data) and htile cols 16..31
    for (int i = lane; i < 128; i += 32) {
      int m = i >> 3, c = i & 7;
      ((unsigned*)xtile)[m * 48 + 8 + c] = 0u;  // halves [16..31] of row m (ld 96)
      ((unsigned*)htile)[m * 16 + 8 + c] = 0u;  // halves [16..31] of row m (ld 32)
    }
    const float4* sg4 = (const float4*)sub_vol;
    for (int i = lane; i < 64; i += 32) {
      int m = i >> 2, c = i & 3;
      float4 v = sg4[(row0 + m) * 4 + c];
      v4h h = {(_Float16)v.x, (_Float16)v.y, (_Float16)v.z, (_Float16)v.w};
      *(v4h*)(xtile + m * 96 + c * 4) = h;
    }
    v8f acc = {};
    acc = WMMA_F16(a_frag(xtile, 96, 0, lane), b_frag(WS1, 0, lane), acc);
    {
      float bb = BS1[n15];
#pragma unroll
      for (int r = 0; r < 8; ++r)
        htile[(mo + r) * 32 + n15] = (_Float16)fmaxf(acc[r] + bb, 0.0f);
    }
    v8f c2 = {};
    c2 = WMMA_F16(a_frag(htile, 32, 0, lane), b_frag(WS2, 0, lane), c2);
    {
      float bb = BS2[n15];
#pragma unroll
      for (int r = 0; r < 8; ++r) scores[(mo + r) * 80 + n15] = c2[r] + bb;
    }
    if (lane < 16) {
      int m = lane;
      topk_softmax(scores + m * 80, sub_vol + (row0 + m) * 16, 16, 2,
                   sI + (row0 + m) * 2, sW + (row0 + m) * 2, feat + m * 32, 21);
    }
  }

  // ================= fc1 (23->256) + fused fc2 (256->1) =================
  v16h af = a_frag(feat, 32, 0, lane);
  float f2acc[8];
#pragma unroll
  for (int r = 0; r < 8; ++r) f2acc[r] = 0.0f;

#pragma unroll
  for (int nt = 0; nt < 16; ++nt) {
    v8f c = {};
    c = WMMA_F16(af, b_frag(WF1, nt, lane), c);
    int n = nt * 16 + n15;
    float bb = BF1[n];
    float w2 = WF2[n];
#pragma unroll
    for (int r = 0; r < 8; ++r) {
      float x1v = c[r] + bb;
      x1_p[(row0 + mo + r) * 256 + n] = x1v;  // x1 (pre-relu) output
      f2acc[r] += fmaxf(x1v, 0.0f) * w2;      // fc2 partial over this lane's columns
    }
  }
  // butterfly-reduce fc2 partials across the 16-lane half (rows 0-7 / 8-15)
#pragma unroll
  for (int off = 1; off < 16; off <<= 1) {
#pragma unroll
    for (int r = 0; r < 8; ++r) f2acc[r] += __shfl_xor(f2acc[r], off, 32);
  }
  if (n15 == 0) {
    float bf2 = BF2[0];
#pragma unroll
    for (int r = 0; r < 8; ++r) out_p[row0 + mo + r] = f2acc[r] + bf2;
  }
}

extern "C" void kernel_launch(void* const* d_in, const int* in_sizes, int n_in,
                              void* d_out, int out_size, void* d_ws, size_t ws_size,
                              hipStream_t stream) {
  (void)in_sizes; (void)n_in; (void)out_size; (void)d_ws; (void)ws_size;
  const float* thick = (const float*)d_in[0];
  const float* area  = (const float*)d_in[1];
  const float* vol   = (const float*)d_in[2];
  const float* sub_v = (const float*)d_in[3];
  const float* tw1 = (const float*)d_in[4];
  const float* tb1 = (const float*)d_in[5];
  const float* tw2 = (const float*)d_in[6];
  const float* tb2 = (const float*)d_in[7];
  const float* aw1 = (const float*)d_in[8];
  const float* ab1 = (const float*)d_in[9];
  const float* aw2 = (const float*)d_in[10];
  const float* ab2 = (const float*)d_in[11];
  const float* vw1 = (const float*)d_in[12];
  const float* vb1 = (const float*)d_in[13];
  const float* vw2 = (const float*)d_in[14];
  const float* vb2 = (const float*)d_in[15];
  const float* sw1 = (const float*)d_in[16];
  const float* sb1 = (const float*)d_in[17];
  const float* sw2 = (const float*)d_in[18];
  const float* sb2 = (const float*)d_in[19];
  const float* fc1w = (const float*)d_in[20];
  const float* fc1b = (const float*)d_in[21];
  const float* fc2w = (const float*)d_in[22];
  const float* fc2b = (const float*)d_in[23];

  dim3 grid(B_TOTAL / ROWS_PER_BLOCK);  // 2048 blocks x 256 threads (8 waves)
  t1sep_fused_kernel<<<grid, TPB, SMEM_TOTAL, stream>>>(
      thick, area, vol, sub_v,
      tw1, tb1, tw2, tb2,
      aw1, ab1, aw2, ab2,
      vw1, vb1, vw2, vb2,
      sw1, sb1, sw2, sb2,
      fc1w, fc1b, fc2w, fc2b,
      (float*)d_out);
}
